// RON_16381005267399
// MI455X (gfx1250) — matrix operand order repaired and checked by reading, not compiled, not run on hardware
//
#include <hip/hip_runtime.h>
#include <hip/hip_bf16.h>

typedef __attribute__((ext_vector_type(16))) _Float16 v16h;
typedef __attribute__((ext_vector_type(8)))  _Float16 v8h;
typedef __attribute__((ext_vector_type(8)))  float    v8f;

#define RON_DT 0.042f
#define RB 128
#define RL 512
#define RNI 96
#define RNH 1024

// ---------------------------------------------------------------------------
// Prep: convert/transpose weights to fp16 fragment-friendly layouts, convert x
// to fp16, zero the recurrent state.
//   h2hT[n][k] = h2h[k][n]   (RNH x RNH, fp16)  -> B-fragment rows contiguous in k
//   x2hT[n][k] = x2h[k][n]   (RNH x RNI, fp16)
//   xf = fp16(x)             (RB*RL*RNI)
// ---------------------------------------------------------------------------
__global__ __launch_bounds__(256) void ron_prep(
    const float* __restrict__ x, const float* __restrict__ x2h,
    const float* __restrict__ h2h,
    _Float16* __restrict__ xf, _Float16* __restrict__ x2hT,
    _Float16* __restrict__ h2hT,
    float* __restrict__ hy0, float* __restrict__ hy1,
    _Float16* __restrict__ hyh0, _Float16* __restrict__ hyh1,
    float* __restrict__ hz)
{
    const long idx = (long)blockIdx.x * 256 + threadIdx.x;
    if (idx < (long)RB * RL * RNI) {
        xf[idx] = (_Float16)x[idx];
    }
    if (idx < (long)RNH * RNH) {
        const int n = (int)(idx >> 10);
        const int k = (int)(idx & (RNH - 1));
        h2hT[idx] = (_Float16)h2h[(long)k * RNH + n];
    }
    if (idx < (long)RNH * RNI) {
        const int n = (int)(idx / RNI);
        const int k = (int)(idx % RNI);
        x2hT[idx] = (_Float16)x2h[(long)k * RNH + n];
    }
    if (idx < (long)RB * RNH) {
        hy0[idx] = 0.0f;
        hy1[idx] = 0.0f;
        hyh0[idx] = (_Float16)0.0f;
        hyh1[idx] = (_Float16)0.0f;
        hz[idx] = 0.0f;
    }
}

// ---------------------------------------------------------------------------
// One recurrent timestep:
//   acc[b,h] = hy@h2h + x[:,t,:]@x2h          (WMMA f16 -> f32)
//   s  = tanh(acc + bias)
//   hz' = hz + DT*(s - gamma*hy - eps*hz)
//   hy' = hy + DT*hz'
//
// Wave -> 16x32 output strip (2 x 16x16 tiles, shared A fragment).
// 256 waves total, 4 waves/block, 64 blocks -> ~64 WGPs active per step.
// The dominant K=1024 loop is register-pipelined with a small live set
// (acc 16 + A 16 + B 32 VGPRs) so next-iteration loads stay in flight
// across the current iteration's WMMAs.
//
// Fragment layouts used here (CDNA5 wave32, v_wmma_f32_16x16x32_f16; A/B k order is this code's own):
//   A 16x32 : lane m=lane&15 ; chunk0 k = kb + (lane<16?0:16) .. +7
//                              chunk1 k = chunk0 + 8         (two b128 loads)
//   B 32x16 : lane col n=lane&15 ; 16 contiguous k at kb + (lane<16?0:16)
//             read from transposed weights -> one 32B load
//   C/D     : VGPR r -> row r + (lane<16?0:8), col lane&15
// ---------------------------------------------------------------------------
__global__ __launch_bounds__(128) void ron_step(
    const _Float16* __restrict__ h2hT,     // [RNH][RNH]
    const _Float16* __restrict__ x2hT,     // [RNH][RNI]
    const _Float16* __restrict__ xf,       // [RB][RL][RNI]
    const float* __restrict__ bias,
    const float* __restrict__ gamma,
    const float* __restrict__ epsv,
    const float* __restrict__ hy_cur,      // [RB][RNH] f32 master
    const _Float16* __restrict__ hyh_cur,  // [RB][RNH] f16 GEMM copy
    float* __restrict__ hy_nxt,
    _Float16* __restrict__ hyh_nxt,
    float* __restrict__ hz,                // [RB][RNH] in-place (tile exclusive)
    float* __restrict__ out,               // [RB][RL][RNH]
    int t)
{
    const int lane = threadIdx.x & 31;
    const int wave = threadIdx.x >> 5;
    const int w    = blockIdx.x * 4 + wave;   // 0..255
    const int tm   = w & 7;                   // row tile: b in [tm*16, tm*16+16)
    const int ng   = w >> 3;                  // col group: h in [ng*32, ng*32+32)

    const int  nl = lane & 15;
    const bool hi = lane >= 16;
    const int  a_off = hi ? 16 : 0;           // A chunk base within 32-k block
    const int  b_off = hi ? 16 : 0;           // B k base within 32-k block
    const int  arow  = tm * 16 + nl;          // A-fragment row (batch index)

    v8f acc[2];
#pragma unroll
    for (int j = 0; j < 2; ++j) { v8f z = {}; acc[j] = z; }

    // ---- recurrent part: K = RNH over hy(f16) x h2hT, register-pipelined ----
    const _Float16* aBase = hyh_cur + (long)arow * RNH;
    const _Float16* bCol0 = h2hT + (long)(ng * 32 + nl) * RNH;

    v16h a_cur, b_cur[2];
    {
        v8h a0 = *(const v8h*)(aBase + a_off);
        v8h a1 = *(const v8h*)(aBase + a_off + 8);
        a_cur = __builtin_shufflevector(a0, a1,
                0, 1, 2, 3, 4, 5, 6, 7, 8, 9, 10, 11, 12, 13, 14, 15);
#pragma unroll
        for (int j = 0; j < 2; ++j)
            b_cur[j] = *(const v16h*)(bCol0 + (long)j * 16 * RNH + b_off);
    }

#pragma unroll 2
    for (int kb = 32; kb < RNH; kb += 32) {
        // issue next iteration's loads first (stay in flight across the WMMAs)
        v8h a0 = *(const v8h*)(aBase + kb + a_off);
        v8h a1 = *(const v8h*)(aBase + kb + a_off + 8);
        v16h a_nxt = __builtin_shufflevector(a0, a1,
                     0, 1, 2, 3, 4, 5, 6, 7, 8, 9, 10, 11, 12, 13, 14, 15);
        v16h b_nxt[2];
#pragma unroll
        for (int j = 0; j < 2; ++j)
            b_nxt[j] = *(const v16h*)(bCol0 + (long)j * 16 * RNH + kb + b_off);
        __builtin_prefetch(bCol0 + kb + b_off + 256, 0, 1);

        // compute on the previously loaded fragments
#pragma unroll
        for (int j = 0; j < 2; ++j)
            acc[j] = __builtin_amdgcn_wmma_f32_16x16x32_f16(
                false, a_cur, false, b_cur[j], (short)0, acc[j], false, false);

        a_cur = a_nxt;
#pragma unroll
        for (int j = 0; j < 2; ++j) b_cur[j] = b_nxt[j];
    }
    // drain last pipelined iteration
#pragma unroll
    for (int j = 0; j < 2; ++j)
        acc[j] = __builtin_amdgcn_wmma_f32_16x16x32_f16(
            false, a_cur, false, b_cur[j], (short)0, acc[j], false, false);

    // ---- input part: K = RNI over x[:,t,:](f16) x x2hT (3 iterations) ----
    const _Float16* xBase = xf + ((long)arow * RL + t) * RNI;
    const _Float16* cCol0 = x2hT + (long)(ng * 32 + nl) * RNI;
#pragma unroll
    for (int kb = 0; kb < RNI; kb += 32) {
        v8h a0 = *(const v8h*)(xBase + kb + a_off);
        v8h a1 = *(const v8h*)(xBase + kb + a_off + 8);
        v16h a = __builtin_shufflevector(a0, a1,
                 0, 1, 2, 3, 4, 5, 6, 7, 8, 9, 10, 11, 12, 13, 14, 15);
#pragma unroll
        for (int j = 0; j < 2; ++j) {
            const _Float16* bp = cCol0 + (long)j * 16 * RNI + kb + b_off;
            v16h b = *(const v16h*)bp;
            acc[j] = __builtin_amdgcn_wmma_f32_16x16x32_f16(
                false, a, false, b, (short)0, acc[j], false, false);
        }
    }

    // ---- oscillator state update epilogue ----
#pragma unroll
    for (int j = 0; j < 2; ++j) {
        const int h = ng * 32 + j * 16 + nl;
        const float bv = bias[h];
        const float gv = gamma[h];
        const float ev = epsv[h];
#pragma unroll
        for (int r = 0; r < 8; ++r) {
            const int  b    = tm * 16 + r + (hi ? 8 : 0);
            const long sidx = (long)b * RNH + h;
            const float s   = tanhf(acc[j][r] + bv);
            const float hyo = hy_cur[sidx];
            const float hzo = hz[sidx];
            const float hzn = hzo + RON_DT * (s - gv * hyo - ev * hzo);
            const float hyn = hyo + RON_DT * hzn;
            hz[sidx]      = hzn;
            hy_nxt[sidx]  = hyn;
            hyh_nxt[sidx] = (_Float16)hyn;
            out[((long)b * RL + t) * RNH + h] = hyn;
        }
    }
}

// ---------------------------------------------------------------------------
extern "C" void kernel_launch(void* const* d_in, const int* in_sizes, int n_in,
                              void* d_out, int out_size, void* d_ws, size_t ws_size,
                              hipStream_t stream) {
    (void)in_sizes; (void)n_in; (void)out_size; (void)ws_size;
    const float* x     = (const float*)d_in[0];
    const float* x2h   = (const float*)d_in[1];
    const float* h2h   = (const float*)d_in[2];
    const float* bias  = (const float*)d_in[3];
    const float* gamma = (const float*)d_in[4];
    const float* epsv  = (const float*)d_in[5];
    float* out = (float*)d_out;

    // ---- workspace carve-up (256B aligned), ~16.3 MB total ----
    char* ws = (char*)d_ws;
    size_t off = 0;
    auto take = [&](size_t bytes) -> void* {
        void* p = ws + off;
        off += (bytes + 255) & ~(size_t)255;
        return p;
    };
    _Float16* h2hT = (_Float16*)take((size_t)RNH * RNH * sizeof(_Float16));     // 2 MB
    _Float16* x2hT = (_Float16*)take((size_t)RNH * RNI * sizeof(_Float16));     // 192 KB
    _Float16* xf   = (_Float16*)take((size_t)RB * RL * RNI * sizeof(_Float16)); // 12 MB
    float*    hy0  = (float*)take((size_t)RB * RNH * sizeof(float));            // 512 KB
    float*    hy1  = (float*)take((size_t)RB * RNH * sizeof(float));
    _Float16* hyh0 = (_Float16*)take((size_t)RB * RNH * sizeof(_Float16));      // 256 KB
    _Float16* hyh1 = (_Float16*)take((size_t)RB * RNH * sizeof(_Float16));
    float*    hzb  = (float*)take((size_t)RB * RNH * sizeof(float));            // 512 KB

    // ---- prep: fp16 convert + transpose + zero state ----
    {
        const long total = (long)RB * RL * RNI;   // dominant count
        const int blocks = (int)((total + 255) / 256);
        ron_prep<<<blocks, 256, 0, stream>>>(x, x2h, h2h, xf, x2hT, h2hT,
                                             hy0, hy1, hyh0, hyh1, hzb);
    }

    // ---- 512 recurrent steps, double-buffered hy ----
    float*    hyF[2] = { hy0, hy1 };
    _Float16* hyH[2] = { hyh0, hyh1 };
    for (int t = 0; t < RL; ++t) {
        const int cur = t & 1;
        const int nxt = cur ^ 1;
        ron_step<<<64, 128, 0, stream>>>(h2hT, x2hT, xf, bias, gamma, epsv,
                                         hyF[cur], hyH[cur],
                                         hyF[nxt], hyH[nxt],
                                         hzb, out, t);
    }
}
